// GalileanConv1D_6236292513993
// MI455X (gfx1250) — matrix operand order repaired and checked by reading, not compiled, not run on hardware
//
#include <hip/hip_runtime.h>
#include <stdint.h>

#define B_    16
#define CIN   128
#define LEN   4096
#define COUT  256
#define KS    9
#define PAD   4
#define DT    0.01f
#define REGC  0.01f
#define LDSX  136   // row stride (halves) for x tile [p][c]; keeps 16B alignment for c0 mult of 8

typedef __attribute__((ext_vector_type(16))) __bf16       v16bf;
typedef __attribute__((ext_vector_type(8)))  float        v8f;
typedef __attribute__((ext_vector_type(4)))  unsigned int u32x4;

union FragAB { u32x4 q[2]; v16bf v; };

// round-to-nearest-even f32 -> bf16 (bit twiddle fallback)
__device__ __forceinline__ unsigned short f2bfbits(float f) {
    unsigned u = __builtin_bit_cast(unsigned, f);
    unsigned r = (u + 0x7FFFu + ((u >> 16) & 1u)) >> 16;
    return (unsigned short)r;
}

// pack two f32 -> packed bf16x2; prefer hardware v_cvt_pk_bf16_f32 when available
__device__ __forceinline__ unsigned pack2bf(float lo, float hi) {
#if __has_builtin(__builtin_amdgcn_cvt_pk_bf16_f32)
    typedef __attribute__((ext_vector_type(2))) __bf16 v2bf;
    v2bf r = __builtin_amdgcn_cvt_pk_bf16_f32(lo, hi);
    return __builtin_bit_cast(unsigned, r);
#else
    return (unsigned)f2bfbits(lo) | ((unsigned)f2bfbits(hi) << 16);
#endif
}

// ---------------- prep: weight f32 [o][c][k] -> bf16 [k][o][c] -------------
__global__ void prep_w_kernel(const float* __restrict__ w, __bf16* __restrict__ wbf) {
    int i = blockIdx.x * 256 + threadIdx.x;      // i < 9*256*128 = 294912
    int k   = i >> 15;                           // / (256*128)
    int rem = i & 32767;
    int o = rem >> 7;
    int c = rem & 127;
    unsigned pk = pack2bf(w[o * (CIN * KS) + c * KS + k], 0.f);
    wbf[i] = __builtin_bit_cast(__bf16, (unsigned short)(pk & 0xFFFFu));
}

// ---------------- prep: folded bias + phys scalar --------------------------
__global__ void prep_bias_kernel(const float* __restrict__ w,
                                 const float* __restrict__ bias,
                                 const float* __restrict__ vel,
                                 const float* __restrict__ acc,
                                 float* __restrict__ biasT,
                                 float* __restrict__ phys_out) {
    int o = threadIdx.x;                         // 256 threads, 1 block
    float s = bias[o];
    for (int c = 0; c < CIN; ++c) {
        #pragma unroll
        for (int k = 0; k < KS; ++k) {
            float t = (float)k * DT;
            float off = vel[c * KS + k] * t + 0.5f * acc[c * KS + k] * t * t;
            s += off * w[o * (CIN * KS) + c * KS + k];
        }
    }
    biasT[o] = s;

    float e = 0.f;
    for (int i = o; i < CIN * KS; i += 256) {
        float v = vel[i], a = acc[i];
        e += v * v + a * a;
    }
    __shared__ float red[256];
    red[o] = e;
    __syncthreads();
    for (int st = 128; st > 0; st >>= 1) {
        if (o < st) red[o] += red[o + st];
        __syncthreads();
    }
    if (o == 0) *phys_out = REGC * red[0] / (float)(CIN * KS);
}

// ---------------- main: implicit-GEMM conv via bf16 WMMA -------------------
// block = 256 threads (8 waves); tile = 64 out-channels x 128 positions x 1 batch
// wave w: o-subtile = (w&3)*16 ; l-half = (w>>2)*64, covering 4 l-subtiles (+0,16,32,48)
__global__ void __launch_bounds__(256)
gconv_wmma_kernel(const float*  __restrict__ x,
                  const __bf16* __restrict__ wbf,   // [k][o][c] bf16
                  const float*  __restrict__ biasT, // folded bias [COUT]
                  float*        __restrict__ out) {
    __shared__ __attribute__((aligned(16))) __bf16 xs[136 * LDSX]; // [p][c], p=0..135 (36992 B)

    const int lB  = blockIdx.x * 128;  // position tile base
    const int oB  = blockIdx.y * 64;   // out-channel tile base
    const int b   = blockIdx.z;
    const int tid = threadIdx.x;

    const float* xb = x + (size_t)b * CIN * LEN;

    // ---- Stage x window: positions [lB-4, lB+132), 128 channels, bf16 [p][c] ----
    // thread t owns channel pair (cp, cp+1) and a 34-position quarter of the window.
    {
        const int cp = (tid & 63) * 2;
        const int ph = (tid >> 6) * 34;
        const bool interior = (lB >= PAD) && (lB + 132 <= LEN);  // uniform per block
        if (interior) {
            const float* s0 = xb + (size_t)cp * LEN + (lB - PAD + ph);
            const float* s1 = s0 + LEN;
            #pragma unroll
            for (int j = 0; j < 17; ++j) {
                float2 a = *(const float2*)(s0 + 2 * j);   // positions p, p+1, channel cp
                float2 c = *(const float2*)(s1 + 2 * j);   // positions p, p+1, channel cp+1
                int p = ph + 2 * j;
                *(unsigned*)&xs[(p    ) * LDSX + cp] = pack2bf(a.x, c.x);
                *(unsigned*)&xs[(p + 1) * LDSX + cp] = pack2bf(a.y, c.y);
            }
        } else {
            for (int j = 0; j < 34; ++j) {
                int p = ph + j;
                int g = lB + p - PAD;
                g = g < 0 ? 0 : (g > LEN - 1 ? LEN - 1 : g);
                *(unsigned*)&xs[p * LDSX + cp] =
                    pack2bf(xb[(size_t)cp * LEN + g], xb[(size_t)(cp + 1) * LEN + g]);
            }
        }
    }
    __syncthreads();

    const int wav   = tid >> 5;
    const int lane  = tid & 31;
    const int lrow  = lane & 15;        // row/col within 16
    const int h     = lane >> 4;        // half-wave select
    const int oSub  = (wav & 3) * 16;   // wave's out-channel sub-tile
    const int lHalf = (wav >> 2) * 64;  // wave's l-half; subtiles at +0,16,32,48

    v8f acc[4] = {{}, {}, {}, {}};

    // A rows: wbf[(k*COUT + o)*CIN + c], lane = row (lrow)
    const __bf16* wrow  = wbf + (size_t)(oB + oSub + lrow) * CIN;
    // B cols: lane = col (lrow); K-half offset h*8 folded in
    const __bf16* xrow0 = &xs[(lHalf + lrow) * LDSX + h * 8];

    #pragma unroll
    for (int k = 0; k < KS; ++k) {
        const __bf16* wk = wrow + (size_t)k * COUT * CIN;
        const __bf16* xk = xrow0 + k * LDSX;
        if (k + 1 < KS)  // prefetch next k-plane of weights (global_prefetch_b8)
            __builtin_prefetch(wrow + (size_t)(k + 1) * COUT * CIN, 0, 3);
        #pragma unroll
        for (int c0 = 0; c0 < CIN; c0 += 32) {
            FragAB A;
            // A 16x32 bf16 layout: lane h=0 holds K{0..7,16..23}, h=1 holds K{8..15,24..31}
            A.q[0] = *(const u32x4*)(wk + c0 + h * 8);
            A.q[1] = *(const u32x4*)(wk + c0 + h * 8 + 16);
            #pragma unroll
            for (int s = 0; s < 4; ++s) {
                // B 32x16 bf16: lanes 0-15 hold K{0..7,16..23}, lanes 16-31 hold K{8..15,24..31}
                const __bf16* bp = xk + s * (16 * LDSX) + c0;
                FragAB Bf;
                Bf.q[0] = *(const u32x4*)(bp);
                Bf.q[1] = *(const u32x4*)(bp + 16);
                acc[s] = __builtin_amdgcn_wmma_f32_16x16x32_bf16(false, A.v, false, Bf.v,
                                                                 (short)0, acc[s], false, false);
            }
        }
    }

    // Per-lane folded bias: D vgpr r holds M = r + 8*h
    float bl[8];
    #pragma unroll
    for (int r = 0; r < 8; ++r) bl[r] = biasT[oB + oSub + r + 8 * h];

    // Store: out[b][o][l]; N = lrow; subtiles at lHalf + 16*s
    float* op = out + ((size_t)b * COUT + (oB + oSub + 8 * h)) * LEN + lB + lHalf + lrow;
    #pragma unroll
    for (int s = 0; s < 4; ++s) {
        #pragma unroll
        for (int r = 0; r < 8; ++r) {
            op[(size_t)r * LEN + 16 * s] = acc[s][r] + bl[r];
        }
    }
}

extern "C" void kernel_launch(void* const* d_in, const int* in_sizes, int n_in,
                              void* d_out, int out_size, void* d_ws, size_t ws_size,
                              hipStream_t stream) {
    const float* x    = (const float*)d_in[0];
    const float* w    = (const float*)d_in[1];
    const float* bias = (const float*)d_in[2];
    const float* vel  = (const float*)d_in[3];
    const float* acc  = (const float*)d_in[4];
    // d_in[5..7] = ks, stride, pad (compile-time constants here)

    __bf16* wbf   = (__bf16*)d_ws;                        // 9*256*128 bf16 = 589824 B
    float*  biasT = (float*)((char*)d_ws + 589824);       // 256 f32
    float*  out   = (float*)d_out;
    float*  phys  = out + (size_t)B_ * COUT * LEN;        // last element of d_out

    prep_w_kernel<<<(KS * COUT * CIN) / 256, 256, 0, stream>>>(w, wbf);
    prep_bias_kernel<<<1, 256, 0, stream>>>(w, bias, vel, acc, biasT, phys);

    dim3 grid(LEN / 128, COUT / 64, B_);
    gconv_wmma_kernel<<<grid, 256, 0, stream>>>(x, wbf, biasT, out);
}
